// MinGRU_33809982554176
// MI455X (gfx1250) — matrix operand order repaired and checked by reading, not compiled, not run on hardware
//
#include <hip/hip_runtime.h>
#include <hip/hip_bf16.h>

typedef __attribute__((ext_vector_type(16))) __bf16 v16bf;
typedef __attribute__((ext_vector_type(8)))  float  v8f;
typedef __attribute__((ext_vector_type(4)))  __bf16 bf16x4;

#define B_   8
#define T_   4096
#define DIN  1024
#define D_   1024
#define ROWS (B_*T_)        /* 32768 */
#define NCH  64
#define CL   (T_/NCH)       /* 64 */

// generic LDS pointer -> 32-bit LDS byte offset (low dword of generic addr)
__device__ __forceinline__ unsigned lds_off(const void* p) {
    return (unsigned)(unsigned long long)p;
}

// async memory -> LDS copy, 16 bytes per lane (tracked by ASYNCcnt)
__device__ __forceinline__ void async_copy_b128(unsigned lds_addr, const void* gaddr) {
    asm volatile("global_load_async_to_lds_b128 %0, %1, off"
                 :: "v"(lds_addr), "v"(gaddr) : "memory");
}
__device__ __forceinline__ void wait_asynccnt0() {
    asm volatile("s_wait_asynccnt 0x0" ::: "memory");
}

// ---------------- fp32 -> bf16 convert (X) ----------------
__global__ void k_conv_x(const float* __restrict__ x, __bf16* __restrict__ out, int n4) {
    int i = blockIdx.x * blockDim.x + threadIdx.x;
    if (i >= n4) return;
    float4 v = ((const float4*)x)[i];
    bf16x4 o;
    o.x = (__bf16)v.x; o.y = (__bf16)v.y; o.z = (__bf16)v.z; o.w = (__bf16)v.w;
    ((bf16x4*)out)[i] = o;
}

// -------- transpose+convert weights: Wt[n][k] = bf16(W[k][n]) --------
__global__ void k_conv_w(const float* __restrict__ Wz, const float* __restrict__ Wn,
                         __bf16* __restrict__ Wzt, __bf16* __restrict__ Wnt) {
    int tid = blockIdx.x * blockDim.x + threadIdx.x;   // [0, DIN*D)
    int k = tid & (DIN - 1);
    int n = tid >> 10;
    Wzt[(size_t)n * DIN + k] = (__bf16)Wz[(size_t)k * D_ + n];
    Wnt[(size_t)n * DIN + k] = (__bf16)Wn[(size_t)k * D_ + n];
}

// ---------------- dual-weight bf16 WMMA GEMM + gate epilogue ----------------
// Block tile: 128 rows x 64 cols, computed against BOTH Wz and Wn.
// 8 waves (wave32): wave grid 4(M) x 2(N), each wave does 32x32 per weight.
// Tile staging uses GLOBAL_LOAD_ASYNC_TO_LDS_B128 (no VGPR staging).
__global__ __launch_bounds__(256) void k_gemm(
    const __bf16* __restrict__ Xb,  const __bf16* __restrict__ Wzt,
    const __bf16* __restrict__ Wnt, const float*  __restrict__ bz,
    const float*  __restrict__ bn,  const int*    __restrict__ mask,
    float* __restrict__ aArr, float* __restrict__ bArr)
{
    __shared__ __bf16 lA[128][40];      // 32 K + 8 pad -> 80B row stride (16B aligned)
    __shared__ __bf16 lB[2][64][40];

    const int rowBase = blockIdx.x * 128;
    const int colBase = blockIdx.y * 64;
    const int tid  = threadIdx.x;
    const int wave = tid >> 5;
    const int lane = tid & 31;
    const int wm = (wave & 3) * 32;
    const int wn = (wave >> 2) * 32;
    const int ln = lane & 15;
    const bool lo = (lane < 16);

    // Precompute per-thread staging slots (fixed LDS dst, striding global src).
    // X tile: 512 16B chunks; this thread handles chunks tid and tid+256.
    int xr0 = tid >> 2,          xc0 = (tid & 3) * 8;
    int xr1 = (tid + 256) >> 2,  xc1 = (tid & 3) * 8;  // (tid+256)&3 == tid&3
    unsigned ldsA0 = lds_off(&lA[xr0][xc0]);
    unsigned ldsA1 = lds_off(&lA[xr1][xc1]);
    const __bf16* srcA0 = &Xb[(size_t)(rowBase + xr0) * DIN + xc0];
    const __bf16* srcA1 = &Xb[(size_t)(rowBase + xr1) * DIN + xc1];
    // W tiles: 512 chunks across both weights; chunks tid and tid+256.
    int w0 = tid >> 8,          rem0 = tid & 255;          // w0 == 0
    int w1 = (tid + 256) >> 8,  rem1 = (tid + 256) & 255;  // w1 == 1, rem1 == tid&255... rem1==rem0
    int br0 = rem0 >> 2, bc0 = (rem0 & 3) * 8;
    int br1 = rem1 >> 2, bc1 = (rem1 & 3) * 8;
    unsigned ldsB0 = lds_off(&lB[w0][br0][bc0]);
    unsigned ldsB1 = lds_off(&lB[w1][br1][bc1]);
    const __bf16* srcB0 = &Wzt[(size_t)(colBase + br0) * DIN + bc0];
    const __bf16* srcB1 = &Wnt[(size_t)(colBase + br1) * DIN + bc1];

    v8f acc[2][2][2];
    v8f zero = {0.f,0.f,0.f,0.f,0.f,0.f,0.f,0.f};
    #pragma unroll
    for (int w = 0; w < 2; ++w)
        #pragma unroll
        for (int i = 0; i < 2; ++i)
            #pragma unroll
            for (int j = 0; j < 2; ++j) acc[w][i][j] = zero;

    for (int k0 = 0; k0 < DIN; k0 += 32) {
        __syncthreads();               // previous iteration's fragment reads done
        async_copy_b128(ldsA0, srcA0 + k0);
        async_copy_b128(ldsA1, srcA1 + k0);
        async_copy_b128(ldsB0, srcB0 + k0);
        async_copy_b128(ldsB1, srcB1 + k0);
        wait_asynccnt0();              // this wave's copies landed in LDS
        __syncthreads();               // all waves' copies visible

        union FragU { v16bf v; uint4 u[2]; };
        FragU af[2];
        FragU bfr[2][2];
        // A layout (16-bit, 16x32): lanes 0-15 hold K{0..7,16..23}, lanes 16-31 K{8..15,24..31}
        const int akb = lo ? 0 : 8;
        // B layout (32x16): lanes 0-15 hold K{0..7,16..23}, lanes 16-31 K{8..15,24..31}
        const int bkb = lo ? 0 : 8;
        #pragma unroll
        for (int tm = 0; tm < 2; ++tm) {
            int r = wm + tm * 16 + ln;
            af[tm].u[0] = *(const uint4*)&lA[r][akb];
            af[tm].u[1] = *(const uint4*)&lA[r][akb + 16];
        }
        #pragma unroll
        for (int w = 0; w < 2; ++w)
            #pragma unroll
            for (int tn = 0; tn < 2; ++tn) {
                int r = wn + tn * 16 + ln;
                bfr[w][tn].u[0] = *(const uint4*)&lB[w][r][bkb];
                bfr[w][tn].u[1] = *(const uint4*)&lB[w][r][bkb + 16];
            }
        #pragma unroll
        for (int w = 0; w < 2; ++w)
            #pragma unroll
            for (int tm = 0; tm < 2; ++tm)
                #pragma unroll
                for (int tn = 0; tn < 2; ++tn)
                    acc[w][tm][tn] = __builtin_amdgcn_wmma_f32_16x16x32_bf16(
                        false, af[tm].v, false, bfr[w][tn].v,
                        (short)0, acc[w][tm][tn], false, false);
    }

    // Epilogue: z = sigmoid(accZ+bz), n = tanh(accN+bn); a = z*(1-m); b = (1-z)*n
    // C/D layout: VGPR i -> M = i + (lane<16 ? 0 : 8), N = lane&15
    #pragma unroll
    for (int tm = 0; tm < 2; ++tm)
        #pragma unroll
        for (int tn = 0; tn < 2; ++tn)
            #pragma unroll
            for (int i = 0; i < 8; ++i) {
                int m   = i + (lo ? 0 : 8);
                int row = rowBase + wm + tm * 16 + m;
                int col = colBase + wn + tn * 16 + ln;
                float zv = acc[0][tm][tn][i] + bz[col];
                zv = 1.0f / (1.0f + __expf(-zv));
                float nv = tanhf(acc[1][tm][tn][i] + bn[col]);
                float mf = (float)mask[row];
                size_t o = (size_t)row * D_ + col;
                aArr[o] = zv * (1.0f - mf);
                bArr[o] = (1.0f - zv) * nv;
            }
}

// ---------------- chunked parallel scan ----------------
// Phase A: per-(b,chunk,d) composition of CL affine steps
__global__ void k_scanA(const float* __restrict__ aArr, const float* __restrict__ bArr,
                        float* __restrict__ Ac, float* __restrict__ Bc) {
    int tid = blockIdx.x * blockDim.x + threadIdx.x;   // [0, B*NCH*D)
    int d = tid & (D_ - 1);
    int rest = tid >> 10;
    int c = rest & (NCH - 1);
    int b = rest >> 6;
    size_t base = ((size_t)b * T_ + (size_t)c * CL) * D_ + d;
    float A = 1.0f, Bv = 0.0f;
    for (int t = 0; t < CL; ++t) {
        float at = aArr[base + (size_t)t * D_];
        float bt = bArr[base + (size_t)t * D_];
        Bv = at * Bv + bt;
        A  = at * A;
    }
    size_t o = ((size_t)b * NCH + c) * D_ + d;
    Ac[o] = A; Bc[o] = Bv;
}

// Phase B: scan over chunk aggregates -> h at each chunk start
__global__ void k_scanB(const float* __restrict__ Ac, const float* __restrict__ Bc,
                        const float* __restrict__ h0, float* __restrict__ Hs) {
    int tid = blockIdx.x * blockDim.x + threadIdx.x;   // [0, B*D)
    int d = tid & (D_ - 1);
    int b = tid >> 10;
    float h = h0[(size_t)b * D_ + d];
    for (int c = 0; c < NCH; ++c) {
        size_t o = ((size_t)b * NCH + c) * D_ + d;
        Hs[o] = h;
        h = Ac[o] * h + Bc[o];
    }
}

// Phase C: replay each chunk from its start carry; in-place b -> h in d_out
__global__ void k_scanC(const float* __restrict__ aArr, const float* __restrict__ Hs,
                        float* __restrict__ out, float* __restrict__ fin) {
    int tid = blockIdx.x * blockDim.x + threadIdx.x;   // [0, B*NCH*D)
    int d = tid & (D_ - 1);
    int rest = tid >> 10;
    int c = rest & (NCH - 1);
    int b = rest >> 6;
    float h = Hs[((size_t)b * NCH + c) * D_ + d];
    size_t base = ((size_t)b * T_ + (size_t)c * CL) * D_ + d;
    for (int t = 0; t < CL; ++t) {
        size_t o = base + (size_t)t * D_;
        float at = aArr[o];
        float bt = out[o];     // b stored here by GEMM epilogue
        h = at * h + bt;
        out[o] = h;
    }
    if (c == NCH - 1) fin[(size_t)b * D_ + d] = h;
}

extern "C" void kernel_launch(void* const* d_in, const int* in_sizes, int n_in,
                              void* d_out, int out_size, void* d_ws, size_t ws_size,
                              hipStream_t stream) {
    const float* inputs = (const float*)d_in[0];
    const int*   mask   = (const int*)  d_in[1];
    const float* h0     = (const float*)d_in[2];
    const float* Wz     = (const float*)d_in[3];
    const float* bz     = (const float*)d_in[4];
    const float* Wn     = (const float*)d_in[5];
    const float* bn     = (const float*)d_in[6];

    char* w = (char*)d_ws;
    __bf16* Xb   = (__bf16*)w;  w += (size_t)ROWS * DIN * 2;   // 67.1 MB
    __bf16* Wzt  = (__bf16*)w;  w += (size_t)DIN * D_ * 2;     // 2 MB
    __bf16* Wnt  = (__bf16*)w;  w += (size_t)DIN * D_ * 2;     // 2 MB
    float*  aArr = (float*)w;   w += (size_t)ROWS * D_ * 4;    // 134.2 MB
    float*  Ac   = (float*)w;   w += (size_t)B_ * NCH * D_ * 4;
    float*  Bc   = (float*)w;   w += (size_t)B_ * NCH * D_ * 4;
    float*  Hs   = (float*)w;

    float* bArr = (float*)d_out;                    // b values, scanned in place
    float* fin  = (float*)d_out + (size_t)ROWS * D_;

    k_conv_x<<<(ROWS * DIN / 4 + 255) / 256, 256, 0, stream>>>(inputs, Xb, ROWS * DIN / 4);
    k_conv_w<<<(DIN * D_) / 256, 256, 0, stream>>>(Wz, Wn, Wzt, Wnt);
    k_gemm<<<dim3(ROWS / 128, D_ / 64), 256, 0, stream>>>(Xb, Wzt, Wnt, bz, bn, mask, aArr, bArr);
    k_scanA<<<(B_ * NCH * D_) / 256, 256, 0, stream>>>(aArr, bArr, Ac, Bc);
    k_scanB<<<(B_ * D_) / 256, 256, 0, stream>>>(Ac, Bc, h0, Hs);
    k_scanC<<<(B_ * NCH * D_) / 256, 256, 0, stream>>>(aArr, Hs, bArr, fin);
}
